// Encoder_10582799417535
// MI455X (gfx1250) — matrix operand order repaired and checked by reading, not compiled, not run on hardware
//
#include <hip/hip_runtime.h>

typedef __attribute__((ext_vector_type(16))) _Float16 v16h;
typedef __attribute__((ext_vector_type(8)))  float    v8f;

#define B_   8
#define N_   1024
#define K_   32
#define H_   128
#define NEGV (-1e30f)

// ---------------- workspace layout (bytes) ----------------
#define WS_WC   0
#define WS_US   (4u*1024u*1024u)
#define WS_PK   (8u*1024u*1024u)
#define WS_NLL  (WS_PK + 7u*32768u + 65536u)

// halves of scan-weight block (bcW1, bcW2, lpW2, lpW1 contiguous in pk)
#define SCANW_HALVES (3 * 16384 + 32768)

// =====================================================================
// Weight packing: B-fragment layout for v_wmma_f32_16x16x32_f16.
// frag (kb,nb): lane L holds N = nb*16+(L&15); half e is K = kb*32+8*(L>>4)+(e<8?e:e+8).
// =====================================================================
__global__ void pack_weights_kernel(const float* w0, const float* w1, const float* w2,
                                    const float* w3, const float* w4, const float* w5,
                                    const float* w6, const float* w7lp, _Float16* dst) {
  int tid = blockIdx.x * blockDim.x + threadIdx.x;
  const int total = 7 * 16384 + 32768;
  for (int idx = tid; idx < total; idx += gridDim.x * blockDim.x) {
    const float* src;
    int local;
    if (idx < 7 * 16384) {
      int m = idx / 16384; local = idx % 16384;
      const float* srcs[7] = {w0, w1, w2, w3, w4, w5, w6};
      src = srcs[m];
    } else { src = w7lp; local = idx - 7 * 16384; }
    int e    = local & 15;
    int lane = (local >> 4) & 31;
    int fb   = local >> 9;     // kb*8+nb
    int nb   = fb & 7;
    int kb   = fb >> 3;
    dst[idx] = (_Float16)src[(kb * 32 + (lane >> 4) * 8 + (e < 8 ? e : e + 8)) * H_ + nb * 16 + (lane & 15)];
  }
}

// A-fragment K index for element e of v16h, given lane group (ISA 16-bit A layout)
__device__ __forceinline__ int a_kidx(int e, int grp) {
  int j = e >> 1, hh = e & 1;
  return (j < 4) ? (j * 2 + hh + grp * 8) : (16 + (j - 4) * 2 + hh + grp * 8);
}

// =====================================================================
// Front: wc = mlp2(X, wc-params), us = mlp2(X, us-params).
// =====================================================================
__global__ __launch_bounds__(256)
void front_mlp2_kernel(const float* X, const _Float16* pk,
                       const float* wc_b1, const float* wc_a1, const float* wc_b2,
                       const float* us_b1, const float* us_a1, const float* us_b2,
                       float* wcOut, float* usOut) {
  __shared__ float xt[16 * H_];
  __shared__ __align__(16) _Float16 hbh[16 * H_];
  int tid = threadIdx.x, lane = tid & 31, wave = tid >> 5;
  int grp = lane >> 4, Ml = lane & 15;
  long r0 = (long)blockIdx.x * 16;
  for (int i = tid; i < 16 * H_; i += 256) xt[i] = X[r0 * H_ + i];
  __syncthreads();

  const _Float16* W1p[2] = {pk + 0 * 16384, pk + 2 * 16384};
  const _Float16* W2p[2] = {pk + 1 * 16384, pk + 3 * 16384};
  const float* b1a[2] = {wc_b1, us_b1};
  const float* b2a[2] = {wc_b2, us_b2};
  const float* a1a[2] = {wc_a1, us_a1};
  float* outa[2] = {wcOut, usOut};
  int nb = wave;
  int col = nb * 16 + Ml;

  for (int p = 0; p < 2; ++p) {
    // layer 1: h = prelu(X@W1 + b1), stored f16 (it only feeds the next WMMA)
    {
      v8f acc = {};
#pragma unroll
      for (int kb = 0; kb < 4; ++kb) {
        v16h a;
#pragma unroll
        for (int e = 0; e < 16; ++e)
          a[e] = (_Float16)xt[Ml * H_ + kb * 32 + a_kidx(e, grp)];
        v16h bf = ((const v16h*)W1p[p])[(kb * 8 + nb) * 32 + lane];
        acc = __builtin_amdgcn_wmma_f32_16x16x32_f16(false, a, false, bf, (short)0, acc, false, false);
      }
      float al = a1a[p][0], bv = b1a[p][col];
#pragma unroll
      for (int r = 0; r < 8; ++r) {
        int row = r + (grp << 3);
        float v = acc[r] + bv;
        hbh[row * H_ + col] = (_Float16)(v > 0.f ? v : al * v);
      }
    }
    __syncthreads();
    // layer 2: out = h@W2 + b2
    {
      v8f acc = {};
#pragma unroll
      for (int kb = 0; kb < 4; ++kb) {
        v16h a;
#pragma unroll
        for (int e = 0; e < 16; ++e)
          a[e] = hbh[Ml * H_ + kb * 32 + a_kidx(e, grp)];
        v16h bf = ((const v16h*)W2p[p])[(kb * 8 + nb) * 32 + lane];
        acc = __builtin_amdgcn_wmma_f32_16x16x32_f16(false, a, false, bf, (short)0, acc, false, false);
      }
      float bv = b2a[p][col];
#pragma unroll
      for (int r = 0; r < 8; ++r) {
        int row = r + (grp << 3);
        outa[p][(r0 + row) * H_ + col] = acc[r] + bv;
      }
    }
    __syncthreads();
  }
}

// =====================================================================
// Scan kernel: one persistent block per batch element.
// =====================================================================
struct AFromBuf {     // f32 source
  const float* buf; int nrows;
  __device__ _Float16 geth(int M, int k) const {
    return (_Float16)((M < nrows) ? buf[M * H_ + k] : 0.f);
  }
};
struct AFromBufH {    // f16 source (hidden activations) — no conversion needed
  const _Float16* buf; int nrows;
  __device__ _Float16 geth(int M, int k) const {
    return (M < nrows) ? buf[M * H_ + k] : (_Float16)0.f;
  }
};
struct AWCPlus {      // rows<33: WC+wc_i ; row 33: wc_i ; rows>=34: 0
  const float* WC; const float* rowI;
  __device__ _Float16 geth(int M, int k) const {
    float v = (M < 33) ? WC[M * H_ + k] : 0.f;
    if (M <= 33) v += rowI[k];
    return (_Float16)v;
  }
};
struct ACat {         // [US | BC] concat, 256 wide
  const float* US; const float* BC;
  __device__ _Float16 geth(int M, int k) const {
    float v = (k < H_) ? US[k] : ((M < 34) ? BC[M * H_ + (k - H_)] : 0.f);
    return (_Float16)v;
  }
};

// One MLP layer over 48 padded rows (34 written): 24 tiles across 8 waves.
// OH=1 -> write f16 to outh ; OH=0 -> write f32 to out.
// combine=1: out[row] = S - bc_base(out[row]) + acc  (row 33: S + acc)
template <class AB, int NKB, int OH>
__device__ void mlp_layer(const AB& ab, const _Float16* Wp, const float* bias,
                          float alpha, int doPrelu, float* out, _Float16* outh, int outRows,
                          const float* S, int combine, int wave, int lane) {
  int grp = lane >> 4, Ml = lane & 15;
  for (int t = wave; t < 24; t += 8) {
    int m0 = (t >> 3) << 4;
    int nb = t & 7;
    v8f acc = {};
#pragma unroll
    for (int kb = 0; kb < NKB; ++kb) {
      v16h a;
#pragma unroll
      for (int e = 0; e < 16; ++e)
        a[e] = ab.geth(m0 + Ml, kb * 32 + a_kidx(e, grp));
      v16h bf = ((const v16h*)Wp)[(kb * 8 + nb) * 32 + lane];
      acc = __builtin_amdgcn_wmma_f32_16x16x32_f16(false, a, false, bf, (short)0, acc, false, false);
    }
    int col = nb * 16 + Ml;
    float bv = bias[col];
#pragma unroll
    for (int r = 0; r < 8; ++r) {
      int row = m0 + r + (grp << 3);
      if (row < outRows) {
        float v = acc[r] + bv;
        if (doPrelu) v = v > 0.f ? v : alpha * v;
        if (OH) {
          outh[row * H_ + col] = (_Float16)v;
        } else {
          if (combine) v = (row < 33) ? (S[col] - out[row * H_ + col] + v) : (S[col] + v);
          out[row * H_ + col] = v;
        }
      }
    }
  }
}

struct ScanSmem {
  // LDS-resident packed weights: bcW1 @0, bcW2 @16384, lpW2 @32768, lpW1 @49152
  __align__(32) _Float16 wpk[SCANW_HALVES];   // 160 KB
  __align__(16) _Float16 bufAh[34 * H_];      // f16 hidden activations
  float WC[33 * H_];
  float bufB[34 * H_];
  float US[H_], S[H_];
  float rowI[H_], rowIm1[H_], usRow[H_];
  float w3[H_];
  float bbc1[H_], bbc2[H_], blp1[H_], blp2[H_];
  float lpart[34 * 4];
  float logits[34];
  int counts[N_];
  int seen[64];
  int nt, idxT, yi, lastf;
};

__global__ __launch_bounds__(256)
void scan_kernel(const float* wcAll, const float* usAll, const _Float16* pk, const int* y,
                 const float* bc_b1, const float* bc_a1, const float* bc_b2,
                 const float* lp_b1, const float* lp_a1, const float* lp_b2,
                 const float* lp_a2, const float* lp_W3, float* nllPartial) {
  __shared__ ScanSmem sm;
  int tid = threadIdx.x, lane = tid & 31, wave = tid >> 5;
  int b = blockIdx.x;
  const float* wcG = wcAll + (long)b * N_ * H_;
  const float* usG = usAll + (long)b * N_ * H_;

  // stage all four scan weight matrices into LDS (contiguous in pk from m4)
  {
    const float4* src = (const float4*)(pk + 4 * 16384);  // 8 halves per float4
    float4* dstv = (float4*)sm.wpk;
    for (int i2 = tid; i2 < SCANW_HALVES / 8; i2 += 256) dstv[i2] = src[i2];
  }
  const _Float16* bcW1p = sm.wpk + 0;
  const _Float16* bcW2p = sm.wpk + 16384;
  const _Float16* lpW2p = sm.wpk + 32768;
  const _Float16* lpW1p = sm.wpk + 49152;

  for (int i2 = tid; i2 < 33 * H_; i2 += 256) sm.WC[i2] = 0.f;
  if (tid < H_) {  // cache small params in LDS (off the per-step critical path)
    sm.w3[tid]   = lp_W3[tid];
    sm.bbc1[tid] = bc_b1[tid];
    sm.bbc2[tid] = bc_b2[tid];
    sm.blp1[tid] = lp_b1[tid];
    sm.blp2[tid] = lp_b2[tid];
  }
  __syncthreads();
  if (tid < H_) {
    sm.WC[tid] = wcG[tid];             // WC0 row 0 = wc[b,0]
    float s = 0.f;
    for (int n = 2; n < N_; ++n) s += usG[n * H_ + tid];   // US0
    sm.US[tid] = s;
  }
  if (tid == 0) {                       // counts = cumsum(first-occurrence)
    for (int v = 0; v < 64; ++v) sm.seen[v] = 0;
    int c = 0;
    for (int n = 0; n < N_; ++n) {
      int lbl = y[n] & 63;
      if (!sm.seen[lbl]) { sm.seen[lbl] = 1; ++c; }
      sm.counts[n] = c;
    }
  }
  float nll = 0.f;
  float bcA1 = bc_a1[0], lpA1 = lp_a1[0], lpA2 = lp_a2[0];
  __syncthreads();

  for (int i = 2; i < N_; ++i) {
    if (tid < H_) {
      sm.rowI[tid]   = wcG[i * H_ + tid];
      sm.rowIm1[tid] = wcG[(i - 1) * H_ + tid];
      sm.usRow[tid]  = usG[i * H_ + tid];
    }
    if (tid == 0) {
      int nt = sm.counts[i - 1], pv = sm.counts[i - 2];
      sm.nt = nt;
      sm.idxT = (nt == pv) ? y[i - 1] : (nt - 1);
      sm.yi = y[i];
      sm.lastf = (i == N_ - 1) ? 1 : 0;
    }
    __syncthreads();
    if (tid < H_) {                      // state updates (reference step order)
      sm.WC[sm.idxT * H_ + tid] += sm.rowIm1[tid];
      sm.US[tid] = sm.lastf ? 0.f : (sm.US[tid] - sm.usRow[tid]);
    }
    __syncthreads();

    // bc_base = mlp2(WC)
    { AFromBuf ab{sm.WC, 33};
      mlp_layer<AFromBuf, 4, 1>(ab, bcW1p, sm.bbc1, bcA1, 1, nullptr, sm.bufAh, 34, nullptr, 0, wave, lane); }
    __syncthreads();
    { AFromBufH ab{sm.bufAh, 34};
      mlp_layer<AFromBufH, 4, 0>(ab, bcW2p, sm.bbc2, 0.f, 0, sm.bufB, nullptr, 34, nullptr, 0, wave, lane); }
    __syncthreads();
    if (tid < H_) {                      // S = sum of active slots
      float s = 0.f; int nt = sm.nt;
      for (int k = 0; k < nt; ++k) s += sm.bufB[k * H_ + tid];
      sm.S[tid] = s;
    }
    __syncthreads();

    // bc_plus = mlp2(WC + wc_i); row33 = mlp2(wc_i); fuse BC_k = S - base + plus
    { AWCPlus ab{sm.WC, sm.rowI};
      mlp_layer<AWCPlus, 4, 1>(ab, bcW1p, sm.bbc1, bcA1, 1, nullptr, sm.bufAh, 34, nullptr, 0, wave, lane); }
    __syncthreads();
    { AFromBufH ab{sm.bufAh, 34};
      mlp_layer<AFromBufH, 4, 0>(ab, bcW2p, sm.bbc2, 0.f, 0, sm.bufB, nullptr, 34, sm.S, 1, wave, lane); }
    __syncthreads();

    // lp = mlp3([US | BC]) ; 34 rows (33 slots + "new")
    { ACat ab{sm.US, sm.bufB};
      mlp_layer<ACat, 8, 1>(ab, lpW1p, sm.blp1, lpA1, 1, nullptr, sm.bufAh, 34, nullptr, 0, wave, lane); }
    __syncthreads();
    { AFromBufH ab{sm.bufAh, 34};
      mlp_layer<AFromBufH, 4, 0>(ab, lpW2p, sm.blp2, lpA2, 1, sm.bufB, nullptr, 34, nullptr, 0, wave, lane); }
    __syncthreads();
    if (tid < 34 * 4) {                  // final 128 -> 1, 4-way split per row
      int row = tid >> 2, part = tid & 3;
      float s = 0.f;
      for (int k = part * 32; k < part * 32 + 32; ++k)
        s += sm.bufB[row * H_ + k] * sm.w3[k];
      sm.lpart[tid] = s;
    }
    __syncthreads();
    if (tid < 34)
      sm.logits[tid] = sm.lpart[tid * 4] + sm.lpart[tid * 4 + 1] +
                       sm.lpart[tid * 4 + 2] + sm.lpart[tid * 4 + 3];
    __syncthreads();
    if (tid == 0) {                      // masked log-softmax + nll
      int nt = sm.nt, yi = sm.yi;
      float mx = NEGV;
      for (int s2 = 0; s2 <= K_; ++s2) {
        float v = (s2 < nt) ? sm.logits[s2] : ((s2 == nt) ? sm.logits[33] : NEGV);
        mx = fmaxf(mx, v);
      }
      float se = 0.f;
      for (int s2 = 0; s2 <= K_; ++s2) {
        float v = (s2 < nt) ? sm.logits[s2] : ((s2 == nt) ? sm.logits[33] : NEGV);
        se += __expf(v - mx);
      }
      float lse = mx + __logf(se);
      float vy = (yi < nt) ? sm.logits[yi] : ((yi == nt) ? sm.logits[33] : NEGV);
      nll += (lse - vy);
    }
    __syncthreads();
  }
  if (tid == 0) nllPartial[b] = nll;
}

__global__ void finalize_kernel(const float* nllPartial, float* out) {
  if (threadIdx.x == 0 && blockIdx.x == 0) {
    float s = 0.f;
    for (int b = 0; b < B_; ++b) s += nllPartial[b];
    out[0] = s / (float)B_;
  }
}

extern "C" void kernel_launch(void* const* d_in, const int* in_sizes, int n_in,
                              void* d_out, int out_size, void* d_ws, size_t ws_size,
                              hipStream_t stream) {
  (void)in_sizes; (void)n_in; (void)out_size; (void)ws_size;
  const float* X     = (const float*)d_in[0];
  const int*   y     = (const int*)d_in[1];
  const float* wcW1  = (const float*)d_in[2];
  const float* wc_b1 = (const float*)d_in[3];
  const float* wc_a1 = (const float*)d_in[4];
  const float* wcW2  = (const float*)d_in[5];
  const float* wc_b2 = (const float*)d_in[6];
  const float* usW1  = (const float*)d_in[7];
  const float* us_b1 = (const float*)d_in[8];
  const float* us_a1 = (const float*)d_in[9];
  const float* usW2  = (const float*)d_in[10];
  const float* us_b2 = (const float*)d_in[11];
  const float* bcW1  = (const float*)d_in[12];
  const float* bc_b1 = (const float*)d_in[13];
  const float* bc_a1 = (const float*)d_in[14];
  const float* bcW2  = (const float*)d_in[15];
  const float* bc_b2 = (const float*)d_in[16];
  const float* lpW1  = (const float*)d_in[17];
  const float* lp_b1 = (const float*)d_in[18];
  const float* lp_a1 = (const float*)d_in[19];
  const float* lpW2  = (const float*)d_in[20];
  const float* lp_b2 = (const float*)d_in[21];
  const float* lp_a2 = (const float*)d_in[22];
  const float* lpW3  = (const float*)d_in[23];

  char* ws = (char*)d_ws;
  float*     wcOut      = (float*)(ws + WS_WC);
  float*     usOut      = (float*)(ws + WS_US);
  _Float16*  pk         = (_Float16*)(ws + WS_PK);
  float*     nllPartial = (float*)(ws + WS_NLL);

  pack_weights_kernel<<<64, 256, 0, stream>>>(wcW1, wcW2, usW1, usW2, bcW1, bcW2, lpW2, lpW1, pk);
  front_mlp2_kernel<<<(B_ * N_) / 16, 256, 0, stream>>>(X, pk, wc_b1, wc_a1, wc_b2,
                                                        us_b1, us_a1, us_b2, wcOut, usOut);
  scan_kernel<<<B_, 256, 0, stream>>>(wcOut, usOut, pk, y, bc_b1, bc_a1, bc_b2,
                                      lp_b1, lp_a1, lp_b2, lp_a2, lpW3, nllPartial);
  finalize_kernel<<<1, 64, 0, stream>>>(nllPartial, (float*)d_out);
}
